// HybridQLSTM_65481071400230
// MI455X (gfx1250) — matrix operand order repaired and checked by reading, not compiled, not run on hardware
//
#include <hip/hip_runtime.h>
#include <hip/hip_bf16.h>

// ---------------- problem constants ----------------
constexpr int TSEQ  = 4096;
constexpr int EDIM  = 1024;
constexpr int HDIM  = 1024;
constexpr int VOCABN = 50257;          // tagset / output vocab
constexpr int VPAD   = 50304;          // 3144*16 = 786*64, padded for 64-row super-tiles
constexpr int GROWS  = 4 * HDIM;       // 4096 stacked gate rows [f;i;g;o]

// ---------------- workspace layout (bytes) ----------------
constexpr size_t WXB_OFF = 0;                                        // bf16 [4096][1024]
constexpr size_t WHB_OFF = WXB_OFF + (size_t)GROWS * EDIM * 2;       // bf16 [4096][1024]
constexpr size_t WTB_OFF = WHB_OFF + (size_t)GROWS * HDIM * 2;       // bf16 [50304][1024]
constexpr size_t XB_OFF  = WTB_OFF + (size_t)VPAD * HDIM * 2;        // bf16 [4096][1024]
constexpr size_t HB_OFF  = XB_OFF  + (size_t)TSEQ * EDIM * 2;        // bf16 [4096][1024]
constexpr size_t PRE_OFF = HB_OFF  + (size_t)TSEQ * HDIM * 2;        // f32  [4096][4096]

// ---------------- types ----------------
typedef __bf16 bf16_t;
typedef __attribute__((ext_vector_type(16))) __bf16 v16bf;
typedef __attribute__((ext_vector_type(8)))  __bf16 v8bf;
typedef __attribute__((ext_vector_type(8)))  float  v8f;

// ---------------- WMMA fragment helpers (wave32; contiguous k order, same in A and B) ----
// A (16x32 bf16, row-major source, row stride ldA):
//   lane L: row m = L&15; lanes 0-15 hold K 0..15, lanes 16-31 hold K 16..31
__device__ __forceinline__ v16bf load_a_frag(const bf16_t* __restrict__ Arow0,
                                             int ldA, int k0, int lane) {
  const int m  = lane & 15;
  const int hi = lane >> 4;
  const bf16_t* p = Arow0 + (size_t)m * ldA + k0 + hi * 16;
  union { v16bf v; v8bf h[2]; } u;
  u.h[0] = *(const v8bf*)(p);        // K = k0 + hi*16 + 0..7
  u.h[1] = *(const v8bf*)(p + 8);    // K = k0 + hi*16 + 8..15
  return u.v;
}

// B (32x16 bf16) with B[k][n] = act[n0+n][k0+k], act row-major with stride ldB:
//   lane L: col n = L&15; lanes 0-15 hold K k0..k0+15, lanes 16-31 hold k0+16..k0+31,
//   ascending in packed element order -> one contiguous 32-byte load.
__device__ __forceinline__ v16bf load_b_frag(const bf16_t* __restrict__ B,
                                             int ldB, int n0, int k0, int lane) {
  const int n  = lane & 15;
  const int hi = lane >> 4;
  return *(const v16bf*)(B + (size_t)(n0 + n) * ldB + k0 + hi * 16);
}

// Broadcast B: B[k][n] = h[k0+k] for every column n (matvec trick).
__device__ __forceinline__ v16bf load_b_bcast(const bf16_t* h, int k0, int lane) {
  const int hi = lane >> 4;
  return *(const v16bf*)(h + k0 + hi * 16);
}

__device__ __forceinline__ v8f wmma_bf16(v16bf a, v16bf b, v8f c) {
  return __builtin_amdgcn_wmma_f32_16x16x32_bf16(false, a, false, b, (short)0, c,
                                                 false, false);
}

__device__ __forceinline__ float sigmoidf_fast(float x) {
  return 1.0f / (1.0f + __expf(-x));
}

// ---------------- weight / activation packing ----------------
__global__ __launch_bounds__(256) void pack_gate_weights(
    const float* __restrict__ Wf, const float* __restrict__ Wi,
    const float* __restrict__ Wg, const float* __restrict__ Wo,
    bf16_t* __restrict__ Wxb, bf16_t* __restrict__ Whb) {
  const size_t per_gate = (size_t)HDIM * (EDIM + HDIM);
  size_t idx = (size_t)blockIdx.x * 256 + threadIdx.x;
  if (idx >= 4 * per_gate) return;
  const int g = (int)(idx / per_gate);
  const size_t rem = idx % per_gate;
  const int r = (int)(rem / (EDIM + HDIM));
  const int c = (int)(rem % (EDIM + HDIM));
  const float* W = (g == 0) ? Wf : (g == 1) ? Wi : (g == 2) ? Wg : Wo;
  const float v = W[rem];
  const int row = g * HDIM + r;
  if (c < EDIM) Wxb[(size_t)row * EDIM + c] = (bf16_t)v;
  else          Whb[(size_t)row * HDIM + (c - EDIM)] = (bf16_t)v;
}

__global__ __launch_bounds__(256) void pack_wt(const float* __restrict__ Wt,
                                               bf16_t* __restrict__ Wtb) {
  size_t idx = (size_t)blockIdx.x * 256 + threadIdx.x;
  if (idx >= (size_t)VPAD * HDIM) return;
  const int v = (int)(idx / HDIM);
  const int k = (int)(idx % HDIM);
  Wtb[idx] = (v < VOCABN) ? (bf16_t)Wt[(size_t)v * HDIM + k] : (bf16_t)0.0f;
}

__global__ __launch_bounds__(256) void gather_embed(const int* __restrict__ sent,
                                                    const float* __restrict__ emb,
                                                    bf16_t* __restrict__ xb) {
  size_t idx = (size_t)blockIdx.x * 256 + threadIdx.x;
  if (idx >= (size_t)TSEQ * EDIM) return;
  const int t = (int)(idx / EDIM);
  const int e = (int)(idx % EDIM);
  xb[idx] = (bf16_t)emb[(size_t)sent[t] * EDIM + e];
}

// ---------------- register-blocked WMMA GEMM ----------------
// D[n][m] = sum_k A[m][k] * B[n][k] (+bias[m]); each wave owns a 64x32 macro-tile
// (4 M-tiles x 2 N-tiles): 8 WMMAs per K-chunk from 4 A-frags + 2 B-frags.
__global__ __launch_bounds__(256) void wmma_gemm_bias(
    const bf16_t* __restrict__ A, const bf16_t* __restrict__ B,
    float* __restrict__ D, const float* __restrict__ bias,
    int Mtiles, int Ntiles, int K, int ldA, int ldB, int ldD, int Mvalid) {
  const int lane = threadIdx.x & 31;
  const long wave = ((long)blockIdx.x * blockDim.x + threadIdx.x) >> 5;
  const int MB = Mtiles >> 2;            // 64-row super-tiles
  const int NB = Ntiles >> 1;            // 32-col super-tiles
  const long total = (long)MB * NB;
  if (wave >= total) return;
  const int mtb = (int)(wave % MB);      // consecutive waves share the B super-tile
  const int ntb = (int)(wave / MB);
  const int m0 = mtb * 64, n0 = ntb * 32;

  v8f acc[4][2] = {};
  for (int k0 = 0; k0 < K; k0 += 32) {
    v16bf bfrag0 = load_b_frag(B, ldB, n0, k0, lane);
    v16bf bfrag1 = load_b_frag(B, ldB, n0 + 16, k0, lane);
    if (k0 + 32 < K) {  // gfx1250 global_prefetch_b8, near-cache scope for the streamed A
      __builtin_prefetch(A + (size_t)(m0 + (lane & 15)) * ldA + k0 + 32, 0, 3);
    }
#pragma unroll
    for (int mi = 0; mi < 4; ++mi) {
      v16bf a = load_a_frag(A + (size_t)(m0 + mi * 16) * ldA, ldA, k0, lane);
      acc[mi][0] = wmma_bf16(a, bfrag0, acc[mi][0]);
      acc[mi][1] = wmma_bf16(a, bfrag1, acc[mi][1]);
    }
  }

  const int n = lane & 15, hi = lane >> 4;
#pragma unroll
  for (int ni = 0; ni < 2; ++ni) {
    float* drow = D + (size_t)(n0 + ni * 16 + n) * ldD;
#pragma unroll
    for (int mi = 0; mi < 4; ++mi) {
      const int mbase = m0 + mi * 16 + hi * 8;
#pragma unroll
      for (int r = 0; r < 8; ++r) {
        const int m = mbase + r;
        if (m < Mvalid) drow[m] = acc[mi][ni][r] + (bias ? bias[m] : 0.0f);
      }
    }
  }
}

// ---------------- persistent sequential LSTM (one WGP, 32 waves) ----------------
// gates[t] = pre[t] + Whh @ h_{t-1}; broadcast-B matvec keeps the matrix cores busy.
// Each wave owns 8 of the 256 gate-row tiles, processed as 2 groups of 4 so one
// LDS B-frag feeds 4 back-to-back WMMAs.
__global__ __launch_bounds__(1024) void lstm_seq_kernel(
    const float* __restrict__ pre, const bf16_t* __restrict__ Whb,
    const float* __restrict__ bfv, const float* __restrict__ biv,
    const float* __restrict__ bgv, const float* __restrict__ bov,
    bf16_t* __restrict__ hb) {
  __shared__ __align__(64) bf16_t h_sh[HDIM];
  __shared__ float gates_sh[GROWS];
  const int tid  = threadIdx.x;          // 0..1023
  const int lane = tid & 31;
  const int wave = tid >> 5;             // 0..31
  float c_reg = 0.0f;
  h_sh[tid] = (bf16_t)0.0f;
  __syncthreads();

  for (int t = 0; t < TSEQ; ++t) {
    // --- Whh @ h via WMMA ---
#pragma unroll
    for (int grp = 0; grp < 2; ++grp) {
      v8f acc[4] = {};
      for (int k0 = 0; k0 < HDIM; k0 += 32) {
        v16bf b = load_b_bcast(h_sh, k0, lane);
#pragma unroll
        for (int i = 0; i < 4; ++i) {
          const int row0 = (wave + (grp * 4 + i) * 32) * 16;
          v16bf a = load_a_frag(Whb + (size_t)row0 * HDIM, HDIM, k0, lane);
          acc[i] = wmma_bf16(a, b, acc[i]);
        }
      }
      // every column of D is identical; lanes 0 and 16 publish the 16 rows
      if ((lane & 15) == 0) {
#pragma unroll
        for (int i = 0; i < 4; ++i) {
          const int base = (wave + (grp * 4 + i) * 32) * 16 + (lane >> 4) * 8;
#pragma unroll
          for (int r = 0; r < 8; ++r) gates_sh[base + r] = acc[i][r];
        }
      }
    }
    __syncthreads();

    // --- elementwise gate math: thread j owns hidden unit j, c stays in a VGPR ---
    const int j = tid;
    const float* prt = pre + (size_t)t * GROWS;
    if (t + 1 < TSEQ) __builtin_prefetch(prt + GROWS + j, 0, 0);
    const float fv = sigmoidf_fast(prt[j]            + gates_sh[j]            + bfv[j]);
    const float iv = sigmoidf_fast(prt[HDIM + j]     + gates_sh[HDIM + j]     + biv[j]);
    const float gv = tanhf        (prt[2 * HDIM + j] + gates_sh[2 * HDIM + j] + bgv[j]);
    const float ov = sigmoidf_fast(prt[3 * HDIM + j] + gates_sh[3 * HDIM + j] + bov[j]);
    c_reg = fv * c_reg + iv * gv;
    const float hv = ov * tanhf(c_reg);
    hb[(size_t)t * HDIM + j] = (bf16_t)hv;
    h_sh[j] = (bf16_t)hv;
    __syncthreads();
  }
}

// ---------------- in-place row-wise log-softmax, one block per row ----------------
__global__ __launch_bounds__(256) void logsoftmax_rows(float* __restrict__ out, int V) {
  __shared__ float sm[256], ss[256];
  const int t = blockIdx.x;
  float* row = out + (size_t)t * V;
  float m = -1e30f, s = 0.0f;
  for (int v = threadIdx.x; v < V; v += 256) {
    const float x = row[v];
    const float nm = fmaxf(m, x);
    s = s * __expf(m - nm) + __expf(x - nm);
    m = nm;
  }
  sm[threadIdx.x] = m;
  ss[threadIdx.x] = s;
  __syncthreads();
  for (int off = 128; off > 0; off >>= 1) {
    if (threadIdx.x < off) {
      const float m2 = sm[threadIdx.x + off], s2 = ss[threadIdx.x + off];
      const float nm = fmaxf(sm[threadIdx.x], m2);
      ss[threadIdx.x] = ss[threadIdx.x] * __expf(sm[threadIdx.x] - nm) + s2 * __expf(m2 - nm);
      sm[threadIdx.x] = nm;
    }
    __syncthreads();
  }
  const float lse = sm[0] + __logf(ss[0]);
  for (int v = threadIdx.x; v < V; v += 256) row[v] -= lse;
}

// ---------------- host launcher ----------------
extern "C" void kernel_launch(void* const* d_in, const int* in_sizes, int n_in,
                              void* d_out, int out_size, void* d_ws, size_t ws_size,
                              hipStream_t stream) {
  const int*   sentence = (const int*)  d_in[0];
  const float* emb      = (const float*)d_in[1];
  const float* Wf       = (const float*)d_in[2];
  const float* bfv      = (const float*)d_in[3];
  const float* Wi       = (const float*)d_in[4];
  const float* biv      = (const float*)d_in[5];
  const float* Wg       = (const float*)d_in[6];
  const float* bgv      = (const float*)d_in[7];
  const float* Wo       = (const float*)d_in[8];
  const float* bov      = (const float*)d_in[9];
  const float* Wt       = (const float*)d_in[10];
  const float* bt       = (const float*)d_in[11];
  float* out = (float*)d_out;

  char* ws = (char*)d_ws;
  bf16_t* Wxb = (bf16_t*)(ws + WXB_OFF);
  bf16_t* Whb = (bf16_t*)(ws + WHB_OFF);
  bf16_t* Wtb = (bf16_t*)(ws + WTB_OFF);
  bf16_t* xb  = (bf16_t*)(ws + XB_OFF);
  bf16_t* hb  = (bf16_t*)(ws + HB_OFF);
  float*  pre = (float*) (ws + PRE_OFF);

  // 1) pack weights to bf16 (Wx | Wh split; Wt zero-padded to 50304 rows)
  {
    const size_t n = (size_t)4 * HDIM * (EDIM + HDIM);
    pack_gate_weights<<<(unsigned)((n + 255) / 256), 256, 0, stream>>>(Wf, Wi, Wg, Wo, Wxb, Whb);
  }
  {
    const size_t n = (size_t)VPAD * HDIM;
    pack_wt<<<(unsigned)((n + 255) / 256), 256, 0, stream>>>(Wt, Wtb);
  }
  // 2) embedding gather -> bf16 activations
  {
    const size_t n = (size_t)TSEQ * EDIM;
    gather_embed<<<(unsigned)((n + 255) / 256), 256, 0, stream>>>(sentence, emb, xb);
  }
  // 3) input-side GEMM: pre[T][4H] = x @ Wx^T  (256 M-tiles x 256 N-tiles, 64x32 per wave)
  {
    const int Mtiles = GROWS / 16, Ntiles = TSEQ / 16;
    const long waves = (long)(Mtiles / 4) * (Ntiles / 2);   // 8192
    wmma_gemm_bias<<<(unsigned)(waves / 8), 256, 0, stream>>>(
        Wxb, xb, pre, nullptr, Mtiles, Ntiles, EDIM, EDIM, EDIM, GROWS, GROWS);
  }
  // 4) sequential recurrence (persistent single workgroup, 32 wave32)
  lstm_seq_kernel<<<1, 1024, 0, stream>>>(pre, Whb, bfv, biv, bgv, bov, hb);
  // 5) output GEMM: logits[T][V] = h @ Wt^T + bt  (3144 x 256 tiles, bounds-checked store)
  {
    const int Mtiles = VPAD / 16, Ntiles = TSEQ / 16;       // 3144 x 256
    const long waves = (long)(Mtiles / 4) * (Ntiles / 2);   // 786*128 = 100608
    wmma_gemm_bias<<<(unsigned)(waves / 8), 256, 0, stream>>>(
        Wtb, hb, out, bt, Mtiles, Ntiles, HDIM, HDIM, HDIM, VOCABN, VOCABN);
  }
  // 6) in-place log-softmax per row
  logsoftmax_rows<<<TSEQ, 256, 0, stream>>>(out, VOCABN);
}
